// HybridModel_23398981829258
// MI455X (gfx1250) — compile-verified, matrix operand order checked by reading, not run on hardware
//
#include <hip/hip_runtime.h>
#include <hip/hip_bf16.h>
#include <math.h>

// ---------------------------------------------------------------------------
// Hybrid CNN + 4-qubit VQC for MI455X (gfx1250, wave32).
//
// conv2 (dominant GEMM: M=802816, N=32, K=144->160) runs on the matrix pipe.
// Logical K order is position-major (k_log = tap*16 + channel) so the CDNA5
// A-fragment element (ks, e) maps to:
//     tap  = 2*ks + (e>>3)          (compile-time constant after unroll)
//     chan = 8*(lane>>4) + (e&7)    (8 consecutive channels)
// With conv1 output stored NHWC + zeroed halo [B][16][16][16ch], each A
// half-fragment is one aligned global_load_b128 at an immediate offset:
// no division, no bounds checks, no exec-mask churn in the WMMA loop.
// B fragments are pre-packed (10 KB) in exact per-lane order -> 32B loads.
// ---------------------------------------------------------------------------

typedef __attribute__((ext_vector_type(16))) _Float16 v16h;
typedef __attribute__((ext_vector_type(8)))  _Float16 v8h;
typedef __attribute__((ext_vector_type(8)))  float    v8f;

#define BATCH 4096

// ---------------------------------------------------------------------------
// Kernel 1: pack conv2 weights [32,16,3,3] f32 into per-lane WMMA B-fragments
// f16 [ks(5)][ntile(2)][lane(32)][elem(16)], tap 9 (K pad) zero-filled.
// Element (lane,e): chan = 8*(lane>>4)+(e&7), tap = 2*ks+(e>>3), o = nt*16+(lane&15).
// ---------------------------------------------------------------------------
__global__ void pack_bfrag_kernel(const float* __restrict__ w,
                                  _Float16* __restrict__ bf) {
    int idx = blockIdx.x * blockDim.x + threadIdx.x;
    if (idx >= 5 * 2 * 32 * 16) return;
    int e = idx & 15;
    int t = idx >> 4;
    int lane = t & 31; t >>= 5;
    int nt = t & 1;
    int ks = t >> 1;
    int ic  = ((lane >> 4) << 3) + (e & 7);
    int pos = 2 * ks + (e >> 3);
    int o   = nt * 16 + (lane & 15);
    float v = 0.f;
    if (pos < 9) v = w[(o * 16 + ic) * 9 + pos];   // conv2_w[o][ic][ky][kx]
    bf[idx] = (_Float16)v;
}

// ---------------------------------------------------------------------------
// Kernel 2: fused conv1(3x3,pad1) + bias + ReLU + 2x2 maxpool, written as
// NHWC f16 with a zeroed 1-pixel halo: h1p[b][Yp(16)][Xp(16)][c(16)],
// interior Yp,Xp in 1..14 <-> pooled coords 0..13.  ReLU(max)==max(ReLU).
// ---------------------------------------------------------------------------
__global__ void conv1_pool_pad_kernel(const float* __restrict__ x,   // [B,1,28,28]
                                      const float* __restrict__ w,   // [16,1,3,3]
                                      const float* __restrict__ bias,
                                      _Float16* __restrict__ h1p) {  // [B,16,16,16]
    int idx = blockIdx.x * blockDim.x + threadIdx.x;
    const int total = BATCH * 16 * 16 * 16;
    if (idx >= total) return;
    int c  = idx & 15; int t = idx >> 4;
    int Xp = t & 15;   t >>= 4;
    int Yp = t & 15;   int b = t >> 4;

    _Float16 res = (_Float16)0.f;                 // halo stays zero
    if (Yp >= 1 && Yp <= 14 && Xp >= 1 && Xp <= 14) {
        int yo = Yp - 1, xo = Xp - 1;
        const float* xb = x + (size_t)b * 28 * 28;
        float in[4][4];
#pragma unroll
        for (int r = 0; r < 4; ++r) {
            int Y = 2 * yo - 1 + r;
#pragma unroll
            for (int q = 0; q < 4; ++q) {
                int X = 2 * xo - 1 + q;
                in[r][q] = (Y >= 0 && Y < 28 && X >= 0 && X < 28) ? xb[Y * 28 + X] : 0.f;
            }
        }
        float wk[9];
#pragma unroll
        for (int i = 0; i < 9; ++i) wk[i] = w[c * 9 + i];
        float bv = bias[c];
        float m = 0.f;                            // ReLU floor
#pragma unroll
        for (int dy = 0; dy < 2; ++dy)
#pragma unroll
            for (int dx = 0; dx < 2; ++dx) {
                float acc = bv;
#pragma unroll
                for (int ky = 0; ky < 3; ++ky)
#pragma unroll
                    for (int kx = 0; kx < 3; ++kx)
                        acc = fmaf(in[dy + ky][dx + kx], wk[ky * 3 + kx], acc);
                m = fmaxf(m, acc);
            }
        res = (_Float16)m;
    }
    h1p[idx] = res;
}

// ---------------------------------------------------------------------------
// Kernel 3: conv2 implicit GEMM on WMMA.  One wave per 16-row M-tile,
// two f32 accumulators for N=0..15 / 16..31, 5 unrolled K=32 steps.
// A half-fragments: aligned b128 loads at immediate offsets from a per-lane
// base into the padded NHWC activation tensor.  Tap 9 (K-pad) -> zero half.
// C/D layout: vgpr r -> row r + 8*(lane>>4), col = lane&15.
// Epilogue fuses bias + ReLU, stores NHWC f16 [B*196, 32].
// ---------------------------------------------------------------------------
__global__ void conv2_wmma_kernel(const _Float16* __restrict__ h1p,   // [B,16,16,16]
                                  const _Float16* __restrict__ bfrag, // [5][2][32][16]
                                  const float* __restrict__ bias,     // [32]
                                  _Float16* __restrict__ out) {       // [B*196, 32]
    const int M = BATCH * 196;
    const int wave = (blockIdx.x * blockDim.x + threadIdx.x) >> 5;
    const int lane = threadIdx.x & 31;
    const int m0 = wave * 16;
    if (m0 >= M) return;                          // uniform per wave

    const int kgrp = (lane >> 4) << 3;            // channel group 0 or 8
    const int m = m0 + (lane & 15);
    const int b = m / 196, sp = m % 196;
    const int y = sp / 14, x = sp % 14;
    // per-lane A base: tap (ky,kx) lives at immediate offset ((ky*16)+kx)*16
    const _Float16* __restrict__ abase =
        h1p + ((((size_t)b * 16 + y) * 16 + x) * 16 + kgrp);
    const _Float16* __restrict__ bbase = bfrag + lane * 16;

    v8f acc0 = {};
    v8f acc1 = {};
#pragma unroll
    for (int ks = 0; ks < 5; ++ks) {
        const int pl = 2 * ks, ph = 2 * ks + 1;   // tap positions (compile-time)
        v8h alo = *(const v8h*)(abase + ((pl / 3) * 16 + (pl % 3)) * 16);
        v8h ahi = {};
        if (ph < 9)                               // tap 9 == K-padding -> zero
            ahi = *(const v8h*)(abase + ((ph / 3) * 16 + (ph % 3)) * 16);
        v16h a = __builtin_shufflevector(alo, ahi,
                 0, 1, 2, 3, 4, 5, 6, 7, 8, 9, 10, 11, 12, 13, 14, 15);
        v16h b0 = *(const v16h*)(bbase + (ks * 2 + 0) * 512);
        v16h b1 = *(const v16h*)(bbase + (ks * 2 + 1) * 512);
        acc0 = __builtin_amdgcn_wmma_f32_16x16x32_f16(
            false, a, false, b0, (short)0, acc0, false, false);
        acc1 = __builtin_amdgcn_wmma_f32_16x16x32_f16(
            false, a, false, b1, (short)0, acc1, false, false);
    }

    const int col = lane & 15;
    const int rb  = (lane >> 4) * 8;
    const float bv0 = bias[col];
    const float bv1 = bias[16 + col];
#pragma unroll
    for (int r = 0; r < 8; ++r) {
        const size_t mm = (size_t)(m0 + rb + r);  // M is a multiple of 16
        out[mm * 32 + col]      = (_Float16)fmaxf(acc0[r] + bv0, 0.f);
        out[mm * 32 + 16 + col] = (_Float16)fmaxf(acc1[r] + bv1, 0.f);
    }
}

// ---------------------------------------------------------------------------
// Kernel 4: 2x2 maxpool on NHWC f16 [B,14,14,32] -> [B,7,7,32]. Coalesced in C.
// ---------------------------------------------------------------------------
__global__ void pool2_kernel(const _Float16* __restrict__ cin,
                             _Float16* __restrict__ pout) {
    int idx = blockIdx.x * blockDim.x + threadIdx.x;
    const int total = BATCH * 7 * 7 * 32;
    if (idx >= total) return;
    int c = idx & 31; int t = idx >> 5;
    int x = t % 7; t /= 7;
    int y = t % 7; int b = t / 7;
    const _Float16* base = cin + (((size_t)b * 14 + 2 * y) * 14 + 2 * x) * 32 + c;
    float m = fmaxf(fmaxf((float)base[0],       (float)base[32]),
                    fmaxf((float)base[14 * 32], (float)base[14 * 32 + 32]));
    pout[idx] = (_Float16)m;
}

// ---------------------------------------------------------------------------
// Kernel 5: FC [B,1568] x [4,1568]^T + bias, sigmoid * pi.  Pooled is NHWC,
// fc_w expects NCHW flatten (c*49 + y*7 + x) -> remap in the index math.
// ---------------------------------------------------------------------------
__global__ void fc_kernel(const _Float16* __restrict__ pooled, // [B,7,7,32]
                          const float* __restrict__ fcw,       // [4,1568]
                          const float* __restrict__ fcb,       // [4]
                          float* __restrict__ feats) {         // [B,4]
    int idx = blockIdx.x * blockDim.x + threadIdx.x;
    if (idx >= BATCH * 4) return;
    int o = idx & 3, b = idx >> 2;
    const _Float16* pb = pooled + (size_t)b * 1568;
    const float* wr = fcw + o * 1568;
    float acc = fcb[o];
    for (int c = 0; c < 32; ++c)
#pragma unroll
        for (int y = 0; y < 7; ++y)
#pragma unroll
            for (int x = 0; x < 7; ++x)
                acc = fmaf((float)pb[(y * 7 + x) * 32 + c], wr[c * 49 + y * 7 + x], acc);
    feats[idx] = 3.14159265358979323846f / (1.f + __expf(-acc));
}

// ---------------------------------------------------------------------------
// Kernel 6: 4-qubit statevector VQC, one thread per sample.
// State index = q0*8 + q1*4 + q2*2 + q3  (wire w <-> bit 8>>w).
// All 16 complex amplitudes live in registers (fully unrolled).
// ---------------------------------------------------------------------------
__device__ __forceinline__ void apply_ry(float* sr, float* si, float th, int bit) {
    float c = __cosf(0.5f * th), s = __sinf(0.5f * th);
#pragma unroll
    for (int i = 0; i < 16; ++i)
        if (!(i & bit)) {
            int j = i | bit;
            float ar = sr[i], ai = si[i], br = sr[j], bi = si[j];
            sr[i] = c * ar - s * br;  si[i] = c * ai - s * bi;
            sr[j] = s * ar + c * br;  si[j] = s * ai + c * bi;
        }
}

__device__ __forceinline__ void apply_rz(float* sr, float* si, float th, int bit) {
    float c = __cosf(0.5f * th), s = __sinf(0.5f * th);
#pragma unroll
    for (int i = 0; i < 16; ++i) {
        float ar = sr[i], ai = si[i];
        if (i & bit) { sr[i] = ar * c - ai * s;  si[i] = ai * c + ar * s; }   // e^{+i th/2}
        else         { sr[i] = ar * c + ai * s;  si[i] = ai * c - ar * s; }   // e^{-i th/2}
    }
}

__global__ void vqc_kernel(const float* __restrict__ feats, // [B,4]
                           const float* __restrict__ wts,   // [3,4,2]
                           float* __restrict__ out) {       // [B,4]
    int b = blockIdx.x * blockDim.x + threadIdx.x;
    if (b >= BATCH) return;
    float sr[16], si[16];
#pragma unroll
    for (int i = 0; i < 16; ++i) { sr[i] = 0.f; si[i] = 0.f; }
    sr[0] = 1.f;

#pragma unroll
    for (int w = 0; w < 4; ++w)
        apply_ry(sr, si, feats[b * 4 + w], 8 >> w);

#pragma unroll
    for (int l = 0; l < 3; ++l) {
        // CNOT ring: (0,1),(1,2),(2,3),(3,0)
#pragma unroll
        for (int w = 0; w < 4; ++w) {
            int cb = 8 >> w, tb = 8 >> ((w + 1) & 3);
#pragma unroll
            for (int i = 0; i < 16; ++i)
                if ((i & cb) && !(i & tb)) {
                    int j = i | tb;
                    float tr = sr[i], ti = si[i];
                    sr[i] = sr[j]; si[i] = si[j];
                    sr[j] = tr;    si[j] = ti;
                }
        }
#pragma unroll
        for (int w = 0; w < 4; ++w) {
            apply_ry(sr, si, wts[(l * 4 + w) * 2 + 0], 8 >> w);
            apply_rz(sr, si, wts[(l * 4 + w) * 2 + 1], 8 >> w);
        }
    }

#pragma unroll
    for (int w = 0; w < 4; ++w) {
        int bit = 8 >> w;
        float e = 0.f;
#pragma unroll
        for (int i = 0; i < 16; ++i) {
            float p = sr[i] * sr[i] + si[i] * si[i];
            e += (i & bit) ? -p : p;
        }
        out[b * 4 + w] = e;
    }
}

// ---------------------------------------------------------------------------
extern "C" void kernel_launch(void* const* d_in, const int* in_sizes, int n_in,
                              void* d_out, int out_size, void* d_ws, size_t ws_size,
                              hipStream_t stream) {
    const float* x   = (const float*)d_in[0];   // [4096,1,28,28]
    const float* w1  = (const float*)d_in[1];   // [16,1,3,3]
    const float* b1  = (const float*)d_in[2];   // [16]
    const float* w2f = (const float*)d_in[3];   // [32,16,3,3]
    const float* b2  = (const float*)d_in[4];   // [32]
    const float* fcw = (const float*)d_in[5];   // [4,1568]
    const float* fcb = (const float*)d_in[6];   // [4]
    const float* vw  = (const float*)d_in[7];   // [3,4,2]
    float* out = (float*)d_out;                 // [4096,4]

    char* ws = (char*)d_ws;
    size_t off = 0;
    _Float16* h1p = (_Float16*)(ws + off); off += (size_t)BATCH * 16 * 16 * 16 * 2; // 33.6 MB
    _Float16* bfr = (_Float16*)(ws + off); off += (size_t)5 * 2 * 32 * 16 * 2;      // 10 KB
    off = (off + 255) & ~(size_t)255;
    _Float16* c2  = (_Float16*)(ws + off); off += (size_t)BATCH * 196 * 32 * 2;     // 51.4 MB
    _Float16* pl  = (_Float16*)(ws + off); off += (size_t)BATCH * 49 * 32 * 2;      // 12.8 MB
    off = (off + 255) & ~(size_t)255;
    float* feats  = (float*)(ws + off);    off += (size_t)BATCH * 4 * 4;

    pack_bfrag_kernel<<<(5 * 2 * 32 * 16 + 255) / 256, 256, 0, stream>>>(w2f, bfr);

    conv1_pool_pad_kernel<<<(BATCH * 16 * 16 * 16 + 255) / 256, 256, 0, stream>>>(
        x, w1, b1, h1p);

    const int waves = (BATCH * 196) / 16;       // 50176 M-tiles
    conv2_wmma_kernel<<<(waves * 32 + 255) / 256, 256, 0, stream>>>(h1p, bfr, b2, c2);

    pool2_kernel<<<(BATCH * 49 * 32 + 255) / 256, 256, 0, stream>>>(c2, pl);

    fc_kernel<<<(BATCH * 4 + 255) / 256, 256, 0, stream>>>(pl, fcw, fcb, feats);

    vqc_kernel<<<(BATCH + 255) / 256, 256, 0, stream>>>(feats, vw, out);
}
